// ScaleAttend_6734508720464
// MI455X (gfx1250) — matrix operand order repaired and checked by reading, not compiled, not run on hardware
//
#include <hip/hip_runtime.h>

typedef __attribute__((ext_vector_type(16))) _Float16 v16h;
typedef __attribute__((ext_vector_type(8)))  _Float16 v8h;
typedef __attribute__((ext_vector_type(4)))  _Float16 v4h;
typedef __attribute__((ext_vector_type(2)))  _Float16 v2h;
typedef __attribute__((ext_vector_type(8)))  float    v8f;

#define WMMA_F16(A, B, C) \
  __builtin_amdgcn_wmma_f32_16x16x32_f16(false, (A), false, (B), (short)0, (C), false, false)

#define CAT16(LO, HI) \
  __builtin_shufflevector((LO), (HI), 0, 1, 2, 3, 4, 5, 6, 7, 8, 9, 10, 11, 12, 13, 14, 15)

// raw v_exp_f32 (2^x) when available; exp2f otherwise
static __device__ __forceinline__ float fast_exp2(float x) {
#if __has_builtin(__builtin_amdgcn_exp2f)
  return __builtin_amdgcn_exp2f(x);
#else
  return exp2f(x);
#endif
}

// Problem shape (fixed by the reference harness)
constexpr int Bb = 4;
constexpr int Hh = 16;
constexpr int Ss = 2048;
constexpr int Dd = 64;
// Tiling
constexpr int KT    = 32;       // keys per main-loop iteration
constexpr int WAVES = 8;        // waves per block (wave32)
constexpr int QBLK  = 128;      // q rows per block (16 per wave)
constexpr int VROW  = KT + 8;   // padded transposed-V row (16B-aligned reads)

__global__ __launch_bounds__(256)
void fa_fwd_wmma(const float* __restrict__ Q,
                 const float* __restrict__ K,
                 const float* __restrict__ V,
                 const unsigned char* __restrict__ Msk,   // bool mask (b, s), 1 byte/elem
                 float* __restrict__ O)
{
  __shared__ __align__(16) _Float16 sKh[KT * Dd];     // 4 KB  f16, (key, d) row-major
  __shared__ __align__(16) _Float16 sVt[Dd * VROW];   // 5 KB  f16, (d, key) transposed+pad
  __shared__ __align__(32) float    sBias[KT];        // per-key additive mask bias (f32)

  const int tid  = threadIdx.x;
  const int lane = tid & 31;
  const int wave = tid >> 5;
  const int col  = lane & 15;   // this lane's q-row within the wave tile
  const int hi   = lane >> 4;   // half-wave select

  const int bh    = blockIdx.x >> 4;   // Ss/QBLK == 16 q-blocks per head
  const int qblk  = blockIdx.x & 15;
  const int batch = bh >> 4;           // Hh == 16

  const float* qb = Q + (size_t)bh * Ss * Dd;
  const float* kb = K + (size_t)bh * Ss * Dd;
  const float* vb = V + (size_t)bh * Ss * Dd;
  const unsigned char* mb = Msk + (size_t)batch * Ss;

  // ---- Q as B-fragments for S^T = K * Q^T (two 32-dim chunks).
  // B layout (16-bit 32x16): lane holds column n = q-row = col; halfs v -> k = 16*hi + v.
  const float  qscale = 0.125f * 1.4426950408889634f;   // (1/sqrt(64)) * log2(e)
  const int    qrow   = qblk * QBLK + wave * 16 + col;
  const float* qp     = qb + (size_t)qrow * Dd;
  v16h bq[2];
  #pragma unroll
  for (int c = 0; c < 2; ++c)
    #pragma unroll
    for (int v = 0; v < 16; ++v)
      bq[c][v] = (_Float16)(qp[c * 32 + 16 * hi + v] * qscale);

  // Transposed output accumulators: O^T, four 16x16 tiles along d.
  // o[dt][r] = O[qrow=col][dim = dt*16 + 8*hi + r]
  v8f o[4] = {};
  float m_ = -1.0e30f;   // per-lane running max (this lane's q-row)
  float l_ = 0.0f;       // per-lane partial denominator (this half's keys)

  // ---- register double-buffer: prologue loads for tile 0 (K, V, mask byte)
  float4 rk0 = ((const float4*)kb)[tid];
  float4 rk1 = ((const float4*)kb)[tid + 256];
  float4 rv0 = ((const float4*)vb)[tid];
  float4 rv1 = ((const float4*)vb)[tid + 256];
  unsigned char mk = 0;
  if (tid < KT) mk = mb[tid];

  const int NT = Ss / KT;   // 64 key tiles
  for (int kt = 0; kt < NT; ++kt) {
    __syncthreads();   // previous tile fully consumed

    // ---- stage registers -> LDS with f32->f16 conversion (once per block)
    {
      v4h h;
      h[0] = (_Float16)rk0.x; h[1] = (_Float16)rk0.y;
      h[2] = (_Float16)rk0.z; h[3] = (_Float16)rk0.w;
      *(v4h*)(sKh + tid * 4) = h;
      h[0] = (_Float16)rk1.x; h[1] = (_Float16)rk1.y;
      h[2] = (_Float16)rk1.z; h[3] = (_Float16)rk1.w;
      *(v4h*)(sKh + (tid + 256) * 4) = h;

      const int f0 = tid * 4,         key0v = f0 >> 6, d0 = f0 & 63;
      const int f1 = (tid + 256) * 4, key1v = f1 >> 6, d1 = f1 & 63;
      sVt[(d0 + 0) * VROW + key0v] = (_Float16)rv0.x;
      sVt[(d0 + 1) * VROW + key0v] = (_Float16)rv0.y;
      sVt[(d0 + 2) * VROW + key0v] = (_Float16)rv0.z;
      sVt[(d0 + 3) * VROW + key0v] = (_Float16)rv0.w;
      sVt[(d1 + 0) * VROW + key1v] = (_Float16)rv1.x;
      sVt[(d1 + 1) * VROW + key1v] = (_Float16)rv1.y;
      sVt[(d1 + 2) * VROW + key1v] = (_Float16)rv1.z;
      sVt[(d1 + 3) * VROW + key1v] = (_Float16)rv1.w;

      if (tid < KT) sBias[tid] = mk ? 0.0f : -1.0e30f;
    }
    __syncthreads();

    // ---- issue next tile's global loads now; they overlap the compute below
    if (kt + 1 < NT) {
      const float4* kn = (const float4*)(kb + (size_t)(kt + 1) * KT * Dd);
      const float4* vn = (const float4*)(vb + (size_t)(kt + 1) * KT * Dd);
      rk0 = kn[tid]; rk1 = kn[tid + 256];
      rv0 = vn[tid]; rv1 = vn[tid + 256];
      if (tid < KT) mk = mb[(kt + 1) * KT + tid];
      if (kt + 2 < NT) {
        __builtin_prefetch(kb + (size_t)(kt + 2) * KT * Dd + tid * 8, 0, 0);
        __builtin_prefetch(vb + (size_t)(kt + 2) * KT * Dd + tid * 8, 0, 0);
      }
    }

    // ---- per-lane bias slices (C-operand of the score WMMAs):
    // st_t[r] belongs to key = kt*32 + t*16 + 8*hi + r -> bias is lane-uniform per key
    const v8f bv0 = *(const v8f*)(sBias + 8 * hi);
    const v8f bv1 = *(const v8f*)(sBias + 16 + 8 * hi);

    // ---- S^T = K * Q^T + bias : 4 independent WMMAs (bias rides in as C)
    // A-frag (K tile, 16 keys x 32 dims): lane row = key = t*16+col;
//   halfs j -> dim c*32 + 16*hi + j.
    v8f t0a, t1a, t0c = {}, t1c = {};
    {
      const _Float16* kr0 = sKh + (col)      * Dd;
      const _Float16* kr1 = sKh + (col + 16) * Dd;
      #define KFRAG(BASE, C) \
        CAT16(*(const v8h*)((BASE) + (C) * 32 + 16 * hi), \
              *(const v8h*)((BASE) + (C) * 32 + 16 * hi + 8))
      t0a = WMMA_F16(KFRAG(kr0, 0), bq[0], bv0);
      t1a = WMMA_F16(KFRAG(kr1, 0), bq[0], bv1);
      t0c = WMMA_F16(KFRAG(kr0, 1), bq[1], t0c);
      t1c = WMMA_F16(KFRAG(kr1, 1), bq[1], t1c);
      #undef KFRAG
    }
    // st_t[r] = masked score(key = kt*32 + t*16 + 8*hi + r, qrow = col)
    const v8f st0 = t0a + t0c;
    const v8f st1 = t1a + t1c;

    // ---- per-lane softmax: this lane owns 16 of its q-row's 32 scores
    float mx;
    {
      float a0 = fmaxf(st0[0], st0[1]), a1 = fmaxf(st0[2], st0[3]);
      float a2 = fmaxf(st0[4], st0[5]), a3 = fmaxf(st0[6], st0[7]);
      float b0 = fmaxf(st1[0], st1[1]), b1 = fmaxf(st1[2], st1[3]);
      float b2 = fmaxf(st1[4], st1[5]), b3 = fmaxf(st1[6], st1[7]);
      a0 = fmaxf(a0, a1); a2 = fmaxf(a2, a3);
      b0 = fmaxf(b0, b1); b2 = fmaxf(b2, b3);
      mx = fmaxf(fmaxf(a0, a2), fmaxf(b0, b2));
      mx = fmaxf(mx, __shfl_xor(mx, 16, 32));   // merge the two half-wave key sets
    }
    const float mn    = fmaxf(m_, mx);
    const float alpha = fast_exp2(m_ - mn);
    m_ = mn;

    float p0[8], p1[8];
    #pragma unroll
    for (int r = 0; r < 8; ++r) {
      p0[r] = fast_exp2(st0[r] - mn);
      p1[r] = fast_exp2(st1[r] - mn);
    }
    const float rs = (((p0[0] + p0[1]) + (p0[2] + p0[3])) + ((p0[4] + p0[5]) + (p0[6] + p0[7])))
                   + (((p1[0] + p1[1]) + (p1[2] + p1[3])) + ((p1[4] + p1[5]) + (p1[6] + p1[7])));
    l_ = l_ * alpha + rs;

    // ---- rescale O^T accumulators with this lane's own alpha (vector pk muls)
    #pragma unroll
    for (int dt = 0; dt < 4; ++dt)
      o[dt] = o[dt] * alpha;

    // ---- assemble P^T B-fragment in registers (half-wave exchange, no LDS):
    // B layout rows k = 16*hi + 0..15; this lane has keys {8hi..8hi+7} U {16+8hi..}.
    v2h pk0[4], pk1[4];
    #pragma unroll
    for (int i = 0; i < 4; ++i) {
      v2h a; a[0] = (_Float16)p0[2 * i]; a[1] = (_Float16)p0[2 * i + 1]; pk0[i] = a;
      v2h b; b[0] = (_Float16)p1[2 * i]; b[1] = (_Float16)p1[2 * i + 1]; pk1[i] = b;
    }
    v16h pb;
    #pragma unroll
    for (int i = 0; i < 4; ++i) {
      const int snd = __builtin_bit_cast(int, hi ? pk0[i] : pk1[i]);
      const v2h rec = __builtin_bit_cast(v2h, __shfl_xor(snd, 16, 32));
      const v2h lo  = hi ? rec    : pk0[i];   // rows 16*hi + 2i, 2i+1
      const v2h hh  = hi ? pk1[i] : rec;      // rows 16*hi + 8 + 2i, 2i+1
      pb[2 * i]     = lo[0]; pb[2 * i + 1]     = lo[1];
      pb[8 + 2 * i] = hh[0]; pb[8 + 2 * i + 1] = hh[1];
    }

    // ---- O^T += V^T * P^T : four independent WMMAs.
    // A-frag (V^T tile, 16 dims x 32 keys): lane row = dim = dt*16+col;
//   halfs j -> key 16*hi + j.
    #pragma unroll
    for (int dt = 0; dt < 4; ++dt) {
      const _Float16* vpd = sVt + (dt * 16 + col) * VROW;
      const v16h av = CAT16(*(const v8h*)(vpd + 16 * hi),
                            *(const v8h*)(vpd + 16 * hi + 8));
      o[dt] = WMMA_F16(av, pb, o[dt]);
    }
  }

  // ---- epilogue: merge half-wave denominators, normalize, contiguous stores
  l_ += __shfl_xor(l_, 16, 32);
  const float inv = 1.0f / fmaxf(l_, 1.0e-30f);
  float* orow = O + ((size_t)bh * Ss + (size_t)(qblk * QBLK + wave * 16 + col)) * Dd;
  #pragma unroll
  for (int dt = 0; dt < 4; ++dt) {
    float4 lo4, hi4;
    lo4.x = o[dt][0] * inv; lo4.y = o[dt][1] * inv;
    lo4.z = o[dt][2] * inv; lo4.w = o[dt][3] * inv;
    hi4.x = o[dt][4] * inv; hi4.y = o[dt][5] * inv;
    hi4.z = o[dt][6] * inv; hi4.w = o[dt][7] * inv;
    *(float4*)(orow + dt * 16 + 8 * hi)     = lo4;
    *(float4*)(orow + dt * 16 + 8 * hi + 4) = hi4;
  }
}

extern "C" void kernel_launch(void* const* d_in, const int* in_sizes, int n_in,
                              void* d_out, int out_size, void* d_ws, size_t ws_size,
                              hipStream_t stream) {
  (void)in_sizes; (void)n_in; (void)out_size; (void)d_ws; (void)ws_size;
  const float*         q   = (const float*)d_in[0];
  const float*         k   = (const float*)d_in[1];
  const float*         v   = (const float*)d_in[2];
  const unsigned char* msk = (const unsigned char*)d_in[3];  // jnp.bool_ -> 1 byte/elem
  float*               out = (float*)d_out;

  const int grid = Bb * Hh * (Ss / QBLK);  // 1024 blocks
  hipLaunchKernelGGL(fa_fwd_wmma, dim3(grid), dim3(256), 0, stream, q, k, v, msk, out);
}
